// MPSClassifier_50062138802239
// MI455X (gfx1250) — matrix operand order repaired and checked by reading, not compiled, not run on hardware
//
#include <hip/hip_runtime.h>

typedef __attribute__((ext_vector_type(16))) _Float16 v16h;
typedef __attribute__((ext_vector_type(8)))  float    v8f;
typedef __attribute__((ext_vector_type(4)))  float    f32x4;

#define BATCH 4096
#define NSITE 256
#define DBOND 128
#define NMID  254
#define NCLS  10
#define SITE_ELEMS 32768          // f16 elements of one pre-swizzled site matrix (256x128)
#define GT_BYTES   ((size_t)NMID * SITE_ELEMS * 2)   // 16,646,144
#define LDS_ROW    264            // 256 f16 + 8 f16 pad (16B-aligned row stride, bank-skewed)

#if __has_builtin(__builtin_amdgcn_sched_barrier)
#define SCHED_FENCE() __builtin_amdgcn_sched_barrier(0)
#else
#define SCHED_FENCE() asm volatile("" ::: "memory")
#endif

// ---------------------------------------------------------------------------
// Prepass 1: convert A_mid (f32, [254][2][128][128]) into f16, pre-swizzled
// into exact WMMA B-fragment order so the main loop's global_load_b128 are
// fully coalesced and stream linearly:
//   GT[m][ ((ks*8 + t)*2 + part)*32 + lane ][e]  =  Acat[m][k'][j]
//   k' = 32*ks + 16*(lane>=16) + 8*part + e ,  j = 16*t + (lane&15)
//   Acat[k'<128] = A_mid[m][0][k'][j],  Acat[k'>=128] = A_mid[m][1][k'-128][j]
// ---------------------------------------------------------------------------
__global__ __launch_bounds__(256) void gt_pack(const float* __restrict__ A_mid,
                                               _Float16* __restrict__ GT) {
    size_t idx = (size_t)blockIdx.x * 256 + threadIdx.x;
    size_t total = (size_t)NMID * SITE_ELEMS;
    if (idx >= total) return;
    int m = (int)(idx / SITE_ELEMS);
    int r = (int)(idx % SITE_ELEMS);
    int e    =  r        & 7;
    int lane = (r >> 3)  & 31;
    int part = (r >> 8)  & 1;
    int t    = (r >> 9)  & 7;
    int ks   = (r >> 12) & 7;
    int hi   = lane >> 4;
    int lcol = lane & 15;
    int kp = 32 * ks + 16 * hi + 8 * part + e;   // 0..255
    int f  = kp >> 7;
    int i  = kp & 127;
    int j  = 16 * t + lcol;
    float v = A_mid[(((size_t)m * 2 + f) * DBOND + i) * DBOND + j];
    GT[idx] = (_Float16)v;
}

// ---------------------------------------------------------------------------
// Prepass 2: feature map, transposed to [site][batch] float2 {cos, sin}.
// ---------------------------------------------------------------------------
__global__ __launch_bounds__(256) void feat_pack(const float* __restrict__ x,
                                                 float2* __restrict__ ft) {
    int idx = blockIdx.x * 256 + threadIdx.x;
    if (idx >= BATCH * NSITE) return;
    int n = idx / BATCH;
    int b = idx % BATCH;
    float a = 1.5707963267948966f * x[(size_t)b * NSITE + n];
    float s, c;
    sincosf(a, &s, &c);
    ft[idx] = make_float2(c, s);
}

// ---------------------------------------------------------------------------
// Main kernel: 64 blocks x 128 threads (4 waves). Wave w owns 16 batch rows.
// Per site: 64x v_wmma_f32_16x16x32_f16, software-pipelined in 16 groups of
// 4 tiles; sched_barrier(0) pins the "load group g+1 / WMMA group g" overlap
// so waits are loadcnt<=8, never 0. Cross-site: group 0 of site m+1 loads
// during the normalize phase; site m+2 swept by global_prefetch_b8.
// ---------------------------------------------------------------------------
__global__ __launch_bounds__(128, 1) void mps_main(const float*  __restrict__ A0,
                                                   const _Float16* __restrict__ GT,
                                                   const float2* __restrict__ ft,
                                                   const float*  __restrict__ AN,
                                                   float* __restrict__ out) {
    __shared__ __align__(16) _Float16 lds[64 * LDS_ROW];   // 33,792 B

    const int tid  = threadIdx.x;
    const int w    = tid >> 5;
    const int lane = tid & 31;
    const int hi   = lane >> 4;
    const int lcol = lane & 15;
    const int gbase = blockIdx.x * 64;

    _Float16* aw = lds + (w * 16) * LDS_ROW;   // wave's 16-row strip

    union AB { v16h h; f32x4 q[2]; };
    v8f acc[8];
    AB  qb[2][4];      // pipelined B fragments: 4 tiles per group
    AB  abuf[2];       // pipelined A fragments (per K-step)

    // group g (0..15): kstep = g>>1, tiles (g&1)*4 .. +3
    auto load_q = [&](AB (&dst)[4], const _Float16* base, int g) {
        int ks = g >> 1, t0 = (g & 1) * 4;
        #pragma unroll
        for (int i = 0; i < 4; ++i) {
            const _Float16* bp = base + (size_t)(ks * 8 + t0 + i) * 512 + lane * 8;
            dst[i].q[0] = *(const f32x4*)(bp);
            dst[i].q[1] = *(const f32x4*)(bp + 256);
        }
    };
    auto load_a = [&](AB& dst, int ks) {
        const _Float16* arow = aw + lcol * LDS_ROW + 32 * ks + 16 * hi;
        dst.q[0] = *(const f32x4*)(arow);        // K 0..7   (hi: 16..23)
        dst.q[1] = *(const f32x4*)(arow + 8);    // K 8..15  (hi: 24..31)
    };

    // kick off site 0 / group 0 B loads immediately (overlaps init below)
    load_q(qb[0], GT, 0);

    // ---- site 0: M0 = cos0*A0[0,:] + sin0*A0[1,:]; prescale with site-1 feats
    {
        int rr   = lcol;
        int grow = gbase + w * 16 + rr;
        float2 f0 = ft[(size_t)0 * BATCH + grow];
        float2 f1 = ft[(size_t)1 * BATCH + grow];
        _Float16* rowp = aw + rr * LDS_ROW;
        for (int j0 = 0; j0 < 64; ++j0) {
            int j = hi * 64 + j0;
            float m0 = f0.x * A0[j] + f0.y * A0[DBOND + j];
            rowp[j]         = (_Float16)(f1.x * m0);
            rowp[DBOND + j] = (_Float16)(f1.y * m0);
        }
    }

    for (int m = 0; m < NMID; ++m) {
        const _Float16* gtm = GT + (size_t)m * SITE_ELEMS;

        #pragma unroll
        for (int t = 0; t < 8; ++t)
            #pragma unroll
            for (int e = 0; e < 8; ++e) acc[t][e] = 0.0f;

        load_a(abuf[0], 0);

        #pragma unroll
        for (int g = 0; g < 16; ++g) {
            const int cur = g & 1;
            const int nxt = cur ^ 1;
            const int ks  = g >> 1;
            const int t0  = cur ? 4 : 0;   // == (g&1)*4

            // ---- issue next group's loads (stay above this group's WMMAs)
            if (g < 15) {
                load_q(qb[nxt], gtm, g + 1);
            } else if (m + 1 < NMID) {
                load_q(qb[nxt], gtm + SITE_ELEMS, 0);  // lands during normalize
            }
            if ((g & 1) == 0 && ks < 7) {
                load_a(abuf[(ks + 1) & 1], ks + 1);    // next K-step A fragment
            }
            if (m + 2 < NMID) {
                // sweep 64KB of site m+2 (full 128B-line coverage per site)
                __builtin_prefetch((const void*)(gtm + 2 * SITE_ELEMS +
                                                 g * 2048 + lane * 64), 0, 1);
            }
            SCHED_FENCE();   // forbid sinking the loads below the WMMAs

            #pragma unroll
            for (int i = 0; i < 4; ++i) {
                acc[t0 + i] = __builtin_amdgcn_wmma_f32_16x16x32_f16(
                    false, abuf[ks & 1].h, false, qb[cur][i].h,
                    (short)0, acc[t0 + i], false, false);
            }
        }

        // ---- row L2 norm: reduce 128 cols per row (8 per-lane + shfl over 16)
        float scl[8];
        #pragma unroll
        for (int v = 0; v < 8; ++v) {
            float s = 0.0f;
            #pragma unroll
            for (int t = 0; t < 8; ++t) { float a_ = acc[t][v]; s += a_ * a_; }
            s += __shfl_xor(s, 1, 32);
            s += __shfl_xor(s, 2, 32);
            s += __shfl_xor(s, 4, 32);
            s += __shfl_xor(s, 8, 32);
            scl[v] = 1.0f / (sqrtf(s) + 1e-8f);
        }

        if (m < NMID - 1) {
            // next site's features, then write scaled f16 A-operand strip
            #pragma unroll
            for (int v = 0; v < 8; ++v) {
                int grow = gbase + w * 16 + v + 8 * hi;
                float2 f = ft[(size_t)(m + 2) * BATCH + grow];
                _Float16* rowp = aw + (v + 8 * hi) * LDS_ROW + lcol;
                float s_ = scl[v];
                #pragma unroll
                for (int t = 0; t < 8; ++t) {
                    float mval = acc[t][v] * s_;
                    rowp[16 * t]         = (_Float16)(f.x * mval);
                    rowp[16 * t + DBOND] = (_Float16)(f.y * mval);
                }
            }
        } else {
            // final normalized M -> f32 LDS (reuse strip, stride 132 floats)
            float* mb = (float*)lds + (w * 16) * 132;
            #pragma unroll
            for (int v = 0; v < 8; ++v) {
                float* rowp = mb + (v + 8 * hi) * 132 + lcol;
                float s_ = scl[v];
                #pragma unroll
                for (int t = 0; t < 8; ++t) rowp[16 * t] = acc[t][v] * s_;
            }
        }
    }

    // ---- epilogue: logits[b,c] = sum_i M[b,i]*(cN*AN0[i,c] + sN*AN1[i,c])
    const float* mb = (const float*)lds + (w * 16) * 132;
    #pragma unroll
    for (int q = 0; q < 5; ++q) {
        int o  = lane + 32 * q;          // 0..159
        int rr = o / NCLS;
        int c  = o % NCLS;
        int grow = gbase + w * 16 + rr;
        float2 fN = ft[(size_t)(NSITE - 1) * BATCH + grow];
        const float* rowp = mb + rr * 132;
        float sum = 0.0f;
        for (int i = 0; i < DBOND; ++i) {
            float wgt = fN.x * AN[i * NCLS + c] + fN.y * AN[DBOND * NCLS + i * NCLS + c];
            sum += rowp[i] * wgt;
        }
        out[(size_t)grow * NCLS + c] = sum;
    }
}

extern "C" void kernel_launch(void* const* d_in, const int* in_sizes, int n_in,
                              void* d_out, int out_size, void* d_ws, size_t ws_size,
                              hipStream_t stream) {
    const float* x     = (const float*)d_in[0];   // [4096,256]
    const float* A0    = (const float*)d_in[1];   // [2,1,128]
    const float* A_mid = (const float*)d_in[2];   // [254,2,128,128]
    const float* AN    = (const float*)d_in[3];   // [2,128,10]
    float* out = (float*)d_out;                   // [4096,10]

    _Float16* GT = (_Float16*)d_ws;                            // 16.6 MB
    float2*   ft = (float2*)((char*)d_ws + GT_BYTES);          // 8 MB

    {
        size_t total = (size_t)NMID * SITE_ELEMS;
        int grid = (int)((total + 255) / 256);
        gt_pack<<<grid, 256, 0, stream>>>(A_mid, GT);
    }
    {
        int total = BATCH * NSITE;
        feat_pack<<<(total + 255) / 256, 256, 0, stream>>>(x, ft);
    }
    mps_main<<<BATCH / 64, 128, 0, stream>>>(A0, GT, ft, AN, out);
}
